// RNNCRF_29016799051947
// MI455X (gfx1250) — matrix operand order repaired and checked by reading, not compiled, not run on hardware
//
#include <hip/hip_runtime.h>
#include <hip/hip_bf16.h>

// BiLSTM-CRF for MI455X (gfx1250, wave32).
// - bf16 WMMA (v_wmma_f32_16x16x32_bf16), f32 accumulate, K software-pipelined x2.
// - GEMM: x tile staged to LDS via TDM tensor_load_to_lds (TENSORcnt), 4x less A traffic.
// - LSTM: gate-interleaved wave ownership -> cell update in registers; per-step xg slab
//   staged via global_load_async_to_lds_b128 (ASYNCcnt), double-buffered one step ahead.
// - xg bf16: 256MB round-trip is the HBM roofline floor (~11us @ 23.3TB/s).

#define B_      64
#define T_      1024
#define HID_    256
#define DIN_PAD 512          // 460 padded to 512 (16 k-chunks of 32)
#define NG      1024         // 4*HID
#define KTAGS   9
#define NEGINF  (-10000.0f)

typedef __bf16 bf16;
typedef __attribute__((ext_vector_type(16))) __bf16 bf16x16;
typedef __attribute__((ext_vector_type(8)))  __bf16 bf16x8;
typedef __attribute__((ext_vector_type(8)))  float  f32x8;
typedef __attribute__((ext_vector_type(4)))  unsigned int u32x4;
typedef __attribute__((ext_vector_type(8)))  unsigned int u32x8;

// ---- A-fragment loader (16x32 bf16, row-major source, ld in elements) ----
// lanes 0-15: M=lane, K=0..15; lanes 16-31: M=lane-16, K=16..31 (matches packed B)
__device__ __forceinline__ bf16x16 load_a_frag(const bf16* base, int ld, int row0, int kt, int lane) {
    int row = row0 + (lane & 15);
    int k0  = kt * 32 + ((lane & 16) ? 16 : 0);
    const bf16* p = base + (size_t)row * ld + k0;
    bf16x8 lo = *(const bf16x8*)(p);
    bf16x8 hi = *(const bf16x8*)(p + 8);
    return __builtin_shufflevector(lo, hi, 0,1,2,3,4,5,6,7,8,9,10,11,12,13,14,15);
}

// ---- B-fragment loader from fragment-major packed buffer (32B/lane) ----
__device__ __forceinline__ bf16x16 load_b_frag(const bf16* pack, int fi, int lane) {
    const bf16* p = pack + ((size_t)fi * 32 + lane) * 16;
    bf16x8 lo = *(const bf16x8*)(p);
    bf16x8 hi = *(const bf16x8*)(p + 8);
    return __builtin_shufflevector(lo, hi, 0,1,2,3,4,5,6,7,8,9,10,11,12,13,14,15);
}

__device__ __forceinline__ f32x8 wmma_bf16(bf16x16 a, bf16x16 b, f32x8 c) {
    return __builtin_amdgcn_wmma_f32_16x16x32_bf16(false, a, false, b, (short)0, c, false, false);
}

__device__ __forceinline__ float sigmoidf_(float x) { return 1.0f / (1.0f + __expf(-x)); }

// ---- CDNA5 async copy: 16B per lane, global -> LDS, tracked by ASYNCcnt ----
__device__ __forceinline__ void async_copy_b128(void* lds_ptr, const void* gptr) {
    unsigned           loff = (unsigned)(unsigned long long)lds_ptr;  // low 32 = LDS offset
    unsigned long long ga   = (unsigned long long)gptr;
    asm volatile("global_load_async_to_lds_b128 %0, %1, off" :: "v"(loff), "v"(ga) : "memory");
}
__device__ __forceinline__ void wait_asynccnt_le8() { asm volatile("s_wait_asynccnt 0x8" ::: "memory"); }
__device__ __forceinline__ void wait_asynccnt_0()   { asm volatile("s_wait_asynccnt 0x0" ::: "memory"); }

// ---- CDNA5 TDM: 1D bf16 tile, global -> LDS, tracked by TENSORcnt ----
// D# per ISA 8.3/8.4: group0={count,lds_addr,global_addr,type=2};
// group1={data_size=1(2B), tensor_dim0=tile_dim0=nelem, dim1=1, stride0=nelem}
__device__ __forceinline__ void tdm_load_1d_b16(unsigned lds_byte_off, const void* gsrc, unsigned nelem) {
    unsigned long long ga = (unsigned long long)gsrc;
    u32x4 g0;
    g0[0] = 1u;                                               // count=1 (valid)
    g0[1] = lds_byte_off;                                     // lds_addr
    g0[2] = (unsigned)(ga & 0xffffffffu);                     // global_addr[31:0]
    g0[3] = (unsigned)((ga >> 32) & 0x01ffffffu) | 0x80000000u; // addr[56:32] | type=2
    u32x8 g1;
    g1[0] = 1u << 16;                                         // data_size=1 -> 2 bytes
    g1[1] = (nelem & 0xffffu) << 16;                          // tensor_dim0[15:0]
    g1[2] = ((nelem >> 16) & 0xffffu) | (1u << 16);           // tensor_dim0[31:16] | tensor_dim1=1
    g1[3] = (nelem & 0xffffu) << 16;                          // tile_dim0
    g1[4] = 1u;                                               // tile_dim1=1, tile_dim2=0
    g1[5] = nelem;                                            // tensor_dim0_stride[31:0]
    g1[6] = 0u;
    g1[7] = 0u;
    asm volatile("tensor_load_to_lds %0, %1" :: "s"(g0), "s"(g1) : "memory");
}

// =====================================================================
// 1) Pack weights into WMMA B-fragment layout (bf16).
//    wih: N=2048 (fwd 0..1023, bwd 1024..2047), K=512 (460 zero-padded)
//    whh: N=2048 (both dirs),                    K=256
//    lpk: N=16   (9 zero-padded),                K=512
// Fragment map: lane<16 -> (N=lane, K=j); lane>=16 -> (N=lane-16, K=16+j)
// =====================================================================
__global__ __launch_bounds__(256) void pack_w_kernel(
    const float* __restrict__ w_ih_f, const float* __restrict__ w_ih_b,
    const float* __restrict__ w_hh_f, const float* __restrict__ w_hh_b,
    const float* __restrict__ lin_w,
    bf16* __restrict__ wih, bf16* __restrict__ whh, bf16* __restrict__ lpk)
{
    const long N1 = 2048L * 512, N2 = 2048L * 256, N3 = 16L * 512;
    long gid = (long)blockIdx.x * 256 + threadIdx.x;
    if (gid < N1) {
        long e = gid;
        int j = (int)(e & 15), lane = (int)((e >> 4) & 31);
        long fi = e >> 9;
        int kt = (int)(fi & 15), nt = (int)(fi >> 4);
        int n = nt * 16 + (lane & 15);
        int k = kt * 32 + ((lane & 16) ? 16 + j : j);
        float v = 0.0f;
        if (k < 460)
            v = (n < 1024) ? w_ih_f[(size_t)n * 460 + k] : w_ih_b[(size_t)(n - 1024) * 460 + k];
        wih[e] = (bf16)v;
    } else if (gid < N1 + N2) {
        long e = gid - N1;
        int j = (int)(e & 15), lane = (int)((e >> 4) & 31);
        long fi = e >> 9;
        int kt = (int)(fi & 7), nt = (int)(fi >> 3);
        int n = nt * 16 + (lane & 15);
        int k = kt * 32 + ((lane & 16) ? 16 + j : j);
        float v = (n < 1024) ? w_hh_f[(size_t)n * 256 + k] : w_hh_b[(size_t)(n - 1024) * 256 + k];
        whh[e] = (bf16)v;
    } else if (gid < N1 + N2 + N3) {
        long e = gid - N1 - N2;
        int j = (int)(e & 15), lane = (int)((e >> 4) & 31);
        int kt = (int)(e >> 9);
        int n = lane & 15;
        int k = kt * 32 + ((lane & 16) ? 16 + j : j);
        float v = (n < KTAGS) ? lin_w[(size_t)n * 512 + k] : 0.0f;
        lpk[e] = (bf16)v;
    }
}

// =====================================================================
// 2) Embedding gather + concat -> bf16 x [B*T, 512]
// =====================================================================
__global__ __launch_bounds__(256) void embed_kernel(
    const int* __restrict__ word_ids, const int* __restrict__ pos_ids,
    const float* __restrict__ vectors,
    const float* __restrict__ word_emb, const float* __restrict__ pos_emb,
    bf16* __restrict__ x)
{
    int r = blockIdx.x;
    int wid = word_ids[r], pid = pos_ids[r];
    for (int c = threadIdx.x; c < DIN_PAD; c += 256) {
        float v;
        if      (c < 128) v = word_emb[(size_t)wid * 128 + c];
        else if (c < 160) v = pos_emb[(size_t)pid * 32 + (c - 128)];
        else if (c < 460) v = vectors[(size_t)r * 300 + (c - 160)];
        else              v = 0.0f;
        x[(size_t)r * DIN_PAD + c] = (bf16)v;
    }
}

// =====================================================================
// 3) xg = x @ [W_ih_f | W_ih_b]^T + bias  ->  bf16 [65536, 2048]
//    WG 256 thr / 8 waves; WG tile 32x512; wave tile 16x128.
//    A tile (32x512 bf16 = 32KB, contiguous) staged to LDS by TDM once;
//    each A row is consumed by 4 waves -> 4x reduction in global A traffic.
//    B fragments double-buffered across K (software pipeline).
// =====================================================================
__global__ __launch_bounds__(256) void gemm_xg_kernel(
    const bf16* __restrict__ x, const bf16* __restrict__ wih,
    const float* __restrict__ b_f, const float* __restrict__ b_b,
    bf16* __restrict__ xg)
{
    __shared__ bf16 xs[32 * DIN_PAD];   // 32KB staged tile
    const int lane = threadIdx.x & 31;
    const int w    = threadIdx.x >> 5;
    const int wm   = w & 1, wn = w >> 1;
    const int mrow0g = blockIdx.x * 32;
    const int ntile0 = blockIdx.y * 32 + wn * 8;

    if (w == 0) {
        tdm_load_1d_b16((unsigned)(unsigned long long)(void*)xs,
                        x + (size_t)mrow0g * DIN_PAD, 32 * DIN_PAD);
        __builtin_amdgcn_s_wait_tensorcnt(0);
    }
    __syncthreads();

    f32x8 acc[8] = {};
    bf16x16 a0, a1, bA[8], bB[8];
    a0 = load_a_frag(xs, DIN_PAD, wm * 16, 0, lane);
    #pragma unroll
    for (int nt = 0; nt < 8; ++nt) bA[nt] = load_b_frag(wih, (ntile0 + nt) * 16, lane);

    #pragma unroll 1
    for (int kt = 0; kt < 16; kt += 2) {
        a1 = load_a_frag(xs, DIN_PAD, wm * 16, kt + 1, lane);
        #pragma unroll
        for (int nt = 0; nt < 8; ++nt)
            bB[nt] = load_b_frag(wih, (ntile0 + nt) * 16 + kt + 1, lane);
        #pragma unroll
        for (int nt = 0; nt < 8; ++nt)
            acc[nt] = wmma_bf16(a0, bA[nt], acc[nt]);
        if (kt < 14) {
            a0 = load_a_frag(xs, DIN_PAD, wm * 16, kt + 2, lane);
            #pragma unroll
            for (int nt = 0; nt < 8; ++nt)
                bA[nt] = load_b_frag(wih, (ntile0 + nt) * 16 + kt + 2, lane);
        }
        #pragma unroll
        for (int nt = 0; nt < 8; ++nt)
            acc[nt] = wmma_bf16(a1, bB[nt], acc[nt]);
    }
    const int m_add = (lane & 16) ? 8 : 0;
    const int col_l = lane & 15;
    #pragma unroll
    for (int nt = 0; nt < 8; ++nt) {
        int col = (ntile0 + nt) * 16 + col_l;
        float bias = (col < NG) ? b_f[col] : b_b[col - NG];
        #pragma unroll
        for (int v = 0; v < 8; ++v) {
            int row = mrow0g + wm * 16 + m_add + v;
            xg[(size_t)row * 2048 + col] = (bf16)(acc[nt][v] + bias);
        }
    }
}

// =====================================================================
// 4) LSTM recurrence. Grid (4 batch-tiles, 2 dirs), 256 thr / 8 waves.
//    Wave w owns units [32w,32w+32): all 4 gates -> c,h update in registers.
//    xg slab (4KB/wave/step) async-staged to LDS one step ahead (ASYNCcnt).
//    W_hh fragments double-buffered across K. Dynamic LDS: 8KB + 2x32KB.
// =====================================================================
__global__ __launch_bounds__(256) void lstm_kernel(
    const bf16* __restrict__ xg, const bf16* __restrict__ whh,
    bf16* __restrict__ hcat)
{
    extern __shared__ char smem[];
    bf16* h_sh  = (bf16*)smem;                 // 16x256 bf16 = 8KB
    char* xgbuf = smem + 16 * 256 * 2;         // 2 x (8 waves x 4KB)

    const int btile = blockIdx.x;
    const int dir   = blockIdx.y;
    const int tid   = threadIdx.x;
    const int lane  = tid & 31;
    const int w     = tid >> 5;

    for (int i = tid; i < 16 * 256; i += 256) h_sh[i] = (bf16)0.0f;
    __syncthreads();

    const int m_add = (lane & 16) ? 8 : 0;
    const int col_l = lane & 15;
    const int fbase = (dir * 64 + w * 2) * 8;   // frag index base for this wave
    // wave's 8 N-tiles: gate g half h -> frag (fbase + (g*16+h)*8 + kt)

    auto issue_xg = [&](int t, int buf) {
        #pragma unroll
        for (int inst = 0; inst < 8; ++inst) {
            int ch   = inst * 32 + lane;               // 256 chunks of 16B
            int row  = ch >> 4;
            int gate = (ch >> 2) & 3;
            int sub  = ch & 3;
            int brow = btile * 16 + row;
            const bf16* gp = xg + ((size_t)brow * T_ + t) * 2048 + dir * NG
                                + gate * 256 + w * 32 + sub * 8;
            void* lp = xgbuf + buf * 32768 + w * 4096 + ch * 16;
            async_copy_b128(lp, gp);
        }
    };

    float creg[2][8] = {};
    issue_xg(dir ? (T_ - 1) : 0, 0);

    for (int s = 0; s < T_; ++s) {
        int t   = dir ? (T_ - 1 - s) : s;
        int buf = s & 1;
        if (s + 1 < T_) issue_xg(dir ? (T_ - 2 - s) : (s + 1), buf ^ 1);

        f32x8 acc[8] = {};
        bf16x16 bA[8], bB[8];
        #pragma unroll
        for (int nt = 0; nt < 8; ++nt) {
            int gate = nt >> 1, half = nt & 1;
            bA[nt] = load_b_frag(whh, fbase + (gate * 16 + half) * 8, lane);
        }
        #pragma unroll 1
        for (int kt = 0; kt < 8; kt += 2) {
            #pragma unroll
            for (int nt = 0; nt < 8; ++nt) {
                int gate = nt >> 1, half = nt & 1;
                bB[nt] = load_b_frag(whh, fbase + (gate * 16 + half) * 8 + kt + 1, lane);
            }
            bf16x16 a = load_a_frag(h_sh, 256, 0, kt, lane);     // ds_load path
            #pragma unroll
            for (int nt = 0; nt < 8; ++nt)
                acc[nt] = wmma_bf16(a, bA[nt], acc[nt]);
            if (kt < 6) {
                #pragma unroll
                for (int nt = 0; nt < 8; ++nt) {
                    int gate = nt >> 1, half = nt & 1;
                    bA[nt] = load_b_frag(whh, fbase + (gate * 16 + half) * 8 + kt + 2, lane);
                }
            }
            a = load_a_frag(h_sh, 256, 0, kt + 1, lane);
            #pragma unroll
            for (int nt = 0; nt < 8; ++nt)
                acc[nt] = wmma_bf16(a, bB[nt], acc[nt]);
        }
        __syncthreads();                    // all waves finished reading h_sh

        if (s + 1 < T_) wait_asynccnt_le8();  // this step's 8 ops landed (in-order)
        else            wait_asynccnt_0();

        const bf16* xt = (const bf16*)(xgbuf + buf * 32768 + w * 4096); // [16][4][32]
        #pragma unroll
        for (int h2 = 0; h2 < 2; ++h2) {
            #pragma unroll
            for (int v = 0; v < 8; ++v) {
                int m  = m_add + v;
                int j2 = h2 * 16 + col_l;
                float gi = sigmoidf_(acc[0 + h2][v] + (float)xt[m * 128 +  0 + j2]);
                float gf = sigmoidf_(acc[2 + h2][v] + (float)xt[m * 128 + 32 + j2]);
                float gg = tanhf    (acc[4 + h2][v] + (float)xt[m * 128 + 64 + j2]);
                float go = sigmoidf_(acc[6 + h2][v] + (float)xt[m * 128 + 96 + j2]);
                float c  = gf * creg[h2][v] + gi * gg;
                creg[h2][v] = c;
                float h  = go * tanhf(c);
                int j    = w * 32 + j2;
                h_sh[m * 256 + j] = (bf16)h;
                int brow = btile * 16 + m;
                hcat[((size_t)brow * T_ + t) * 512 + dir * HID_ + j] = (bf16)h;
            }
        }
        __syncthreads();                    // h_sh ready for next step
    }
}

// =====================================================================
// 5) logits = hcat @ lin_w^T + lin_b  ([65536,512] x [512,16(pad 9)])
// =====================================================================
__global__ __launch_bounds__(256) void logits_kernel(
    const bf16* __restrict__ hcat, const bf16* __restrict__ lpk,
    const float* __restrict__ lin_b, float* __restrict__ out_logits)
{
    const int lane = threadIdx.x & 31;
    const int w    = threadIdx.x >> 5;
    const int mrow0 = (blockIdx.x * 8 + w) * 16;

    f32x8 acc = {};
    #pragma unroll 1
    for (int kt = 0; kt < 16; ++kt) {
        bf16x16 a = load_a_frag(hcat, 512, mrow0, kt, lane);
        bf16x16 b = load_b_frag(lpk, kt, lane);
        acc = wmma_bf16(a, b, acc);
    }
    int col = lane & 15;
    if (col < KTAGS) {
        int m_add = (lane & 16) ? 8 : 0;
        float bias = lin_b[col];
        #pragma unroll
        for (int v = 0; v < 8; ++v) {
            int row = mrow0 + m_add + v;
            out_logits[(size_t)row * KTAGS + col] = acc[v] + bias;
        }
    }
}

// =====================================================================
// 6) Constrained Viterbi (BIO). One thread per sequence (64 threads).
// =====================================================================
__global__ __launch_bounds__(64) void viterbi_kernel(
    const float* __restrict__ logits, const float* __restrict__ trans,
    const float* __restrict__ start_t, const float* __restrict__ end_t,
    unsigned char* __restrict__ bp,
    float* __restrict__ tags_out, float* __restrict__ mask_out)
{
    __shared__ float tc[KTAGS * KTAGS];
    __shared__ float sc[KTAGS];
    __shared__ float ec[KTAGS];
    int tid = threadIdx.x;
    for (int e = tid; e < KTAGS * KTAGS; e += 64) {
        int i = e / KTAGS, j = e % KTAGS;
        bool allowed = (j == 0) || (j & 1) || (i == j - 1) || (i == j);
        tc[e] = allowed ? trans[e] : NEGINF;
    }
    if (tid < KTAGS) {
        sc[tid] = ((tid == 0) || (tid & 1)) ? start_t[tid] : NEGINF;
        ec[tid] = end_t[tid];
    }
    __syncthreads();

    int b = tid;
    float alpha[KTAGS];
    #pragma unroll
    for (int j = 0; j < KTAGS; ++j)
        alpha[j] = sc[j] + logits[((size_t)b * T_) * KTAGS + j];

    for (int t = 1; t < T_; ++t) {
        const float* lg = logits + ((size_t)b * T_ + t) * KTAGS;
        float na[KTAGS];
        #pragma unroll
        for (int j = 0; j < KTAGS; ++j) {
            float best = alpha[0] + tc[j];
            int bi = 0;
            #pragma unroll
            for (int i = 1; i < KTAGS; ++i) {
                float sv = alpha[i] + tc[i * KTAGS + j];
                if (sv > best) { best = sv; bi = i; }
            }
            na[j] = best + lg[j];
            bp[((size_t)b * T_ + t) * KTAGS + j] = (unsigned char)bi;
        }
        #pragma unroll
        for (int j = 0; j < KTAGS; ++j) alpha[j] = na[j];
    }
    float best = alpha[0] + ec[0];
    int cur = 0;
    #pragma unroll
    for (int j = 1; j < KTAGS; ++j) {
        float sv = alpha[j] + ec[j];
        if (sv > best) { best = sv; cur = j; }
    }
    tags_out[(size_t)b * T_ + (T_ - 1)] = (float)cur;
    for (int t = T_ - 1; t >= 1; --t) {
        cur = bp[((size_t)b * T_ + t) * KTAGS + cur];
        tags_out[(size_t)b * T_ + (t - 1)] = (float)cur;
    }
    for (int t = 0; t < T_; ++t) mask_out[(size_t)b * T_ + t] = 1.0f;
}

// =====================================================================
extern "C" void kernel_launch(void* const* d_in, const int* in_sizes, int n_in,
                              void* d_out, int out_size, void* d_ws, size_t ws_size,
                              hipStream_t stream)
{
    (void)in_sizes; (void)n_in; (void)out_size; (void)ws_size;
    const int*   word_ids = (const int*)  d_in[0];
    const int*   pos_ids  = (const int*)  d_in[1];
    const float* vectors  = (const float*)d_in[2];
    /* d_in[3] mask: all ones, unused */
    const float* word_emb = (const float*)d_in[4];
    const float* pos_emb  = (const float*)d_in[5];
    const float* w_ih_f   = (const float*)d_in[6];
    const float* w_hh_f   = (const float*)d_in[7];
    const float* b_f      = (const float*)d_in[8];
    const float* w_ih_b   = (const float*)d_in[9];
    const float* w_hh_b   = (const float*)d_in[10];
    const float* b_b      = (const float*)d_in[11];
    const float* lin_w    = (const float*)d_in[12];
    const float* lin_b    = (const float*)d_in[13];
    const float* trans    = (const float*)d_in[14];
    const float* start_t  = (const float*)d_in[15];
    const float* end_t    = (const float*)d_in[16];

    char* ws = (char*)d_ws;
    bf16*  x_bf = (bf16*) (ws);                           //  64 MB [65536,512]
    bf16*  wih  = (bf16*) (ws + 67108864);                //   2 MB
    bf16*  whh  = (bf16*) (ws + 69206016);                //   1 MB
    bf16*  lpk  = (bf16*) (ws + 70254592);                //  16 KB
    bf16*  xg   = (bf16*) (ws + 70270976);                // 256 MB [65536,2048]
    bf16*  hcat = (bf16*) (ws + 338706432);               //  64 MB [65536,512]
    unsigned char* bp = (unsigned char*)(ws + 405815296); // 576 KB

    float* out_logits = (float*)d_out;                    // [B,T,9]
    float* out_tags   = out_logits + (size_t)B_ * T_ * KTAGS;
    float* out_mask   = out_tags   + (size_t)B_ * T_;

    long npack = 2048L * 512 + 2048L * 256 + 16L * 512;
    pack_w_kernel<<<(int)((npack + 255) / 256), 256, 0, stream>>>(
        w_ih_f, w_ih_b, w_hh_f, w_hh_b, lin_w, wih, whh, lpk);

    embed_kernel<<<B_ * T_, 256, 0, stream>>>(word_ids, pos_ids, vectors,
                                              word_emb, pos_emb, x_bf);

    dim3 g_gemm(2048, 4);
    gemm_xg_kernel<<<g_gemm, 256, 0, stream>>>(x_bf, wih, b_f, b_b, xg);

    dim3 g_lstm(4, 2);
    lstm_kernel<<<g_lstm, 256, 73728 /* 8KB h + 2x32KB xg */, stream>>>(xg, whh, hcat);

    logits_kernel<<<512, 256, 0, stream>>>(hcat, lpk, lin_b, out_logits);

    viterbi_kernel<<<1, 64, 0, stream>>>(out_logits, trans, start_t, end_t,
                                         bp, out_tags, out_mask);
}
